// UniPhyModel_72773925863916
// MI455X (gfx1250) — matrix operand order repaired and checked by reading, not compiled, not run on hardware
//
#include <hip/hip_runtime.h>
#include <hip/hip_bf16.h>

typedef __attribute__((ext_vector_type(16))) _Float16 v16h;
typedef __attribute__((ext_vector_type(8)))  _Float16 v8h;
typedef __attribute__((ext_vector_type(8)))  float    v8f;

#define KB_ 2
#define KT_ 8
#define KC_ 4
#define KD_ 128
#define KF2 256
#define KM_ 4096   /* B*T*Hp*Wp tokens */
#define KE_ 4
#define KHD 1024   /* fused MoE hidden width: E * HD = 4*256 */

__device__ __forceinline__ float sigm(float x) { return 1.f / (1.f + expf(-x)); }
__device__ __forceinline__ float gelu_tanh(float x) {
    const float k0 = 0.7978845608028654f;
    return 0.5f * x * (1.f + tanhf(k0 * (x + 0.044715f * x * x * x)));
}

// A-fragment loader: row-major f16 row pointer, K-offset kb, lane-half lh.
// Fragment layout: lane<16 -> K[0..15]; lane>=16 -> K[16..31]
__device__ __forceinline__ v16h load_afrag(const _Float16* __restrict__ Ar, int kb, int lh) {
    v8h a0 = *(const v8h*)(Ar + kb + lh * 16);
    v8h a1 = *(const v8h*)(Ar + kb + 8 + lh * 16);
    return __builtin_shufflevector(a0, a1, 0, 1, 2, 3, 4, 5, 6, 7,
                                   8, 9, 10, 11, 12, 13, 14, 15);
}

// ---------------------------------------------------------------- prep kernels
__global__ __launch_bounds__(256) void k_prep_cat2T(const float* __restrict__ A,
                                                    const float* __restrict__ B,
                                                    _Float16* __restrict__ dst,
                                                    int K, int N1, int N2) {
    int idx = blockIdx.x * blockDim.x + threadIdx.x;
    int N = N1 + N2;
    if (idx >= K * N) return;
    int n = idx / K, k = idx % K;
    float v = (n < N1) ? A[(size_t)k * N1 + n] : B[(size_t)k * N2 + (n - N1)];
    dst[idx] = (_Float16)v;
}

// Complex mix W = [[Wre, Wim], [-Wim, Wre]] stored col-major f16: dst[n*256+k]
__global__ __launch_bounds__(256) void k_prep_complexT(const float* __restrict__ Wre,
                                                       const float* __restrict__ Wim,
                                                       _Float16* __restrict__ dst) {
    int idx = blockIdx.x * blockDim.x + threadIdx.x;
    if (idx >= 256 * 256) return;
    int n = idx >> 8, k = idx & 255;
    float v;
    if (n < 128) { int d = n;       v = (k < 128) ? Wre[k * 128 + d] : -Wim[(k - 128) * 128 + d]; }
    else         { int d = n - 128; v = (k < 128) ? Wim[k * 128 + d] :  Wre[(k - 128) * 128 + d]; }
    dst[idx] = (_Float16)v;
}

// plain transpose: src[k*N+n] -> dst[n*K+k] f16
__global__ __launch_bounds__(256) void k_prep_T(const float* __restrict__ W,
                                                _Float16* __restrict__ dst, int K, int N) {
    int idx = blockIdx.x * blockDim.x + threadIdx.x;
    if (idx >= K * N) return;
    int n = idx / K, k = idx % K;
    dst[idx] = (_Float16)W[(size_t)k * N + n];
}

// conv_W[o][c][ky][kx] (256,256,3,3) -> dst[tap][n=o][k=c] f16
__global__ __launch_bounds__(256) void k_prep_conv(const float* __restrict__ cw,
                                                   _Float16* __restrict__ dst) {
    int idx = blockIdx.x * blockDim.x + threadIdx.x;
    if (idx >= 9 * 256 * 256) return;
    int tap = idx / (256 * 256);
    int rem = idx % (256 * 256);
    int n = rem >> 8, k = rem & 255;
    dst[idx] = (_Float16)cw[((size_t)n * 256 + k) * 9 + tap];
}

__global__ __launch_bounds__(256) void k_fill0(float* p, int n) {
    int i = blockIdx.x * blockDim.x + threadIdx.x;
    if (i < n) p[i] = 0.f;
}

// ---------------------------------------------------------------- WMMA GEMM
// C[M,N] = A[M,K](f16 row-major, row stride lda) x Bw[N,K](f16 col-major, ld K)
// Macro-tile: 32 rows x 64 cols per wave (2 A-frags x 4 B-frags -> 8 WMMAs/K-step)
enum { EP_STORE = 0, EP_ADD = 1, EP_GELU_F16 = 2, EP_SCALE_STORE = 3, EP_SCALE_ADD = 4 };

__global__ __launch_bounds__(256) void k_gemm(
    const _Float16* __restrict__ A, const _Float16* __restrict__ Bw,
    float* __restrict__ outF, _Float16* __restrict__ outH,
    const float* __restrict__ bias, const float* __restrict__ gates,
    int eidx, int M, int N, int K, int lda, int mode) {
    const int wave = (blockIdx.x * blockDim.x + threadIdx.x) >> 5;
    const int lane = threadIdx.x & 31;
    const int mt = M >> 5, nt = N >> 6;
    if (wave >= mt * nt) return;
    const int tm = wave % mt, tn = wave / mt;
    const int lh = lane >> 4, l15 = lane & 15;
    const _Float16* Ar0 = A + (size_t)(tm * 32 + l15) * lda;
    const _Float16* Ar1 = Ar0 + (size_t)16 * lda;
    const _Float16* Br = Bw + (size_t)(tn * 64 + l15) * K;
    v8f acc[2][4] = {};
    for (int kb = 0; kb < K; kb += 32) {
        v16h a0 = load_afrag(Ar0, kb, lh);
        v16h a1 = load_afrag(Ar1, kb, lh);
#pragma unroll
        for (int j = 0; j < 4; ++j) {
            v16h bv = *(const v16h*)(Br + (size_t)j * 16 * K + kb + lh * 16);
            acc[0][j] = __builtin_amdgcn_wmma_f32_16x16x32_f16(false, a0, false, bv,
                                                               (short)0, acc[0][j], false, false);
            acc[1][j] = __builtin_amdgcn_wmma_f32_16x16x32_f16(false, a1, false, bv,
                                                               (short)0, acc[1][j], false, false);
        }
    }
#pragma unroll
    for (int j = 0; j < 4; ++j) {
        const int col = tn * 64 + j * 16 + l15;
        const float bb = bias ? bias[col] : 0.f;
#pragma unroll
        for (int ii = 0; ii < 2; ++ii) {
#pragma unroll
            for (int r = 0; r < 8; ++r) {
                const int row = tm * 32 + ii * 16 + lh * 8 + r;
                const size_t idx = (size_t)row * N + col;
                float v = acc[ii][j][r] + bb;
                if (mode == EP_STORE)            outF[idx] = v;
                else if (mode == EP_ADD)         outF[idx] += v;
                else if (mode == EP_GELU_F16)    outH[idx] = (_Float16)gelu_tanh(v);
                else if (mode == EP_SCALE_STORE) outF[idx] = gates[(size_t)row * KE_ + eidx] * v;
                else                             outF[idx] += gates[(size_t)row * KE_ + eidx] * v;
            }
        }
    }
}

// 3x3 SAME conv as 9-tap WMMA accumulation; z += conv(xn) + bias
// Macro-tile: 16 rows x 64 cols (A-frag reused by 4 WMMAs; shift predication per row)
__global__ __launch_bounds__(256) void k_conv(
    const _Float16* __restrict__ A /*xn [M][256] f16*/,
    const _Float16* __restrict__ Wt /*[9][n][k] f16*/,
    float* __restrict__ z, const float* __restrict__ bias) {
    const int wave = (blockIdx.x * blockDim.x + threadIdx.x) >> 5;
    const int lane = threadIdx.x & 31;
    const int mt = KM_ >> 4, nt = KF2 >> 6;
    if (wave >= mt * nt) return;
    const int tm = wave % mt, tn = wave / mt;
    const int lh = lane >> 4, l15 = lane & 15;
    const int m = tm * 16 + l15;
    const int wp = m & 15, hp = (m >> 4) & 15, bt = m >> 8;
    v8f acc[4] = {};
    for (int ky = 0; ky < 3; ++ky) {
        const int hp2 = hp + ky - 1;
        for (int kx = 0; kx < 3; ++kx) {
            const int wp2 = wp + kx - 1;
            const bool valid = (hp2 >= 0) && (hp2 < 16) && (wp2 >= 0) && (wp2 < 16);
            const int m2 = ((bt * 16 + hp2) << 4) + wp2;
            const _Float16* Ar = A + (size_t)m2 * KF2;
            const _Float16* Br = Wt + ((size_t)(ky * 3 + kx) * KF2 + tn * 64 + l15) * KF2;
            for (int kb = 0; kb < KF2; kb += 32) {
                v16h av = {};
                if (valid) av = load_afrag(Ar, kb, lh);
#pragma unroll
                for (int j = 0; j < 4; ++j) {
                    v16h bv = *(const v16h*)(Br + (size_t)j * 16 * KF2 + kb + lh * 16);
                    acc[j] = __builtin_amdgcn_wmma_f32_16x16x32_f16(false, av, false, bv,
                                                                    (short)0, acc[j], false, false);
                }
            }
        }
    }
#pragma unroll
    for (int j = 0; j < 4; ++j) {
        const int col = tn * 64 + j * 16 + l15;
#pragma unroll
        for (int r = 0; r < 8; ++r) {
            const int row = tm * 16 + lh * 8 + r;
            z[(size_t)row * KF2 + col] += acc[j][r] + bias[col];
        }
    }
}

// ---------------------------------------------------------------- small kernels
__global__ __launch_bounds__(256) void k_patchify(const float* __restrict__ x,
                                                  _Float16* __restrict__ fA) {
    int idx = blockIdx.x * blockDim.x + threadIdx.x;
    if (idx >= KM_ * 256) return;
    int m = idx >> 8, cpp = idx & 255;
    int wp = m & 15, hp = (m >> 4) & 15, bt = m >> 8;
    int c = cpp >> 6, py = (cpp >> 3) & 7, px = cpp & 7;
    size_t xi = (((size_t)bt * KC_ + c) * 128 + (hp * 8 + py)) * 128 + (wp * 8 + px);
    fA[idx] = (_Float16)x[xi];
}

__global__ __launch_bounds__(256) void k_unpatchify(const float* __restrict__ fo,
                                                    float* __restrict__ out) {
    int idx = blockIdx.x * blockDim.x + threadIdx.x;
    if (idx >= KM_ * 256) return;
    int m = idx >> 8, cpp = idx & 255;
    int wp = m & 15, hp = (m >> 4) & 15, bt = m >> 8;
    int c = cpp >> 6, py = (cpp >> 3) & 7, px = cpp & 7;
    size_t oi = (((size_t)bt * KC_ + c) * 128 + (hp * 8 + py)) * 128 + (wp * 8 + px);
    out[oi] = fo[idx];
}

__global__ __launch_bounds__(256) void k_f32_to_f16(const float* __restrict__ in,
                                                    _Float16* __restrict__ out, int n) {
    int i = blockIdx.x * blockDim.x + threadIdx.x;
    if (i < n) out[i] = (_Float16)in[i];
}

// LayerNorm over 256 channels; optional f16 and f32 outputs.
__global__ __launch_bounds__(256) void k_layernorm(const float* __restrict__ in,
                                                   const float* __restrict__ g,
                                                   const float* __restrict__ b,
                                                   _Float16* outH, float* outF) {
    int m = blockIdx.x, c = threadIdx.x;
    float v = in[(size_t)m * KF2 + c];
    __shared__ float red[256];
    red[c] = v;
    __syncthreads();
    for (int s = 128; s > 0; s >>= 1) { if (c < s) red[c] += red[c + s]; __syncthreads(); }
    float mean = red[0] * (1.f / 256.f);
    __syncthreads();
    float d = v - mean;
    red[c] = d * d;
    __syncthreads();
    for (int s = 128; s > 0; s >>= 1) { if (c < s) red[c] += red[c + s]; __syncthreads(); }
    float var = red[0] * (1.f / 256.f);
    float y = d * rsqrtf(var + 1e-5f) * g[c] + b[c];
    if (outH) outH[(size_t)m * KF2 + c] = (_Float16)y;
    if (outF) outF[(size_t)m * KF2 + c] = y;
}

// xm[bt][ch] = mean over 256 spatial tokens
__global__ __launch_bounds__(256) void k_mean(const float* __restrict__ xe,
                                              float* __restrict__ xm) {
    int bt = blockIdx.x, ch = threadIdx.x;
    float s = 0.f;
    for (int sp = 0; sp < 256; ++sp) s += xe[((size_t)bt * 256 + sp) * KF2 + ch];
    xm[bt * KF2 + ch] = s * (1.f / 256.f);
}

// sequential flux tracker over T (single block)
__global__ __launch_bounds__(256) void k_fluxscan(
    const float* __restrict__ xm, const float* __restrict__ gW,
    const float* __restrict__ gb, const float* __restrict__ sW,
    const float* __restrict__ sb, const float* __restrict__ fdec,
    float* __restrict__ flux, float* __restrict__ fseq,
    float* __restrict__ sseq, float* __restrict__ gbuf) {
    int c = threadIdx.x;
    __shared__ float inp[256];
    __shared__ float gsh[128];
    __shared__ float ssh[256];
    for (int b = 0; b < KB_; ++b) {
        for (int t = 0; t < KT_; ++t) {
            int bt = b * KT_ + t;
            inp[c] = xm[bt * 256 + c];
            __syncthreads();
            float acc = sb[c];
            for (int k = 0; k < 256; ++k) acc += inp[k] * sW[k * 256 + c];
            ssh[c] = acc;
            if (c < 128) {
                float ga = gb[c];
                for (int k = 0; k < 256; ++k) ga += inp[k] * gW[k * 128 + c];
                gsh[c] = sigm(ga);
            }
            __syncthreads();
            if (c < 128) {
                int d = c;
                float dec = sigm(fdec[d]);
                float gg = gsh[d];
                float fr = dec * flux[b * 256 + d] + gg * inp[d] + (1.f - gg) * ssh[d];
                float fi = dec * flux[b * 256 + 128 + d] + gg * inp[128 + d] + (1.f - gg) * ssh[128 + d];
                flux[b * 256 + d] = fr;
                flux[b * 256 + 128 + d] = fi;
                fseq[bt * 256 + d] = fr;
                fseq[bt * 256 + 128 + d] = fi;
                sseq[bt * 256 + d] = ssh[d];
                sseq[bt * 256 + 128 + d] = ssh[128 + d];
            }
            __syncthreads();
        }
    }
    for (int bt = 0; bt < KB_ * KT_; ++bt) {
        inp[c] = fseq[bt * 256 + c];
        __syncthreads();
        if (c < 128) {
            float ga = gb[c];
            for (int k = 0; k < 256; ++k) ga += inp[k] * gW[k * 128 + c];
            gbuf[bt * 128 + c] = sigm(ga);
        }
        __syncthreads();
    }
}

__global__ __launch_bounds__(256) void k_forcing(const float* __restrict__ xe,
                                                 const float* __restrict__ g,
                                                 const float* __restrict__ srcs,
                                                 float* __restrict__ forc) {
    int idx = blockIdx.x * blockDim.x + threadIdx.x;
    if (idx >= KM_ * 256) return;
    int m = idx >> 8, ch = idx & 255;
    int bt = m >> 8, d = ch & 127;
    float gg = g[bt * 128 + d];
    forc[idx] = xe[idx] * gg + srcs[bt * 256 + ch] * (1.f - gg);
}

// ZOH operators opd = exp(dt*lam), opf = (opd-1)/lam, lam = -softplus(lr) + i*li
__global__ __launch_bounds__(256) void k_ops(const float* __restrict__ dt,
                                             const float* __restrict__ lam_re,
                                             const float* __restrict__ lam_im,
                                             float* odr_, float* odi_, float* ofr_, float* ofi_) {
    int idx = blockIdx.x * blockDim.x + threadIdx.x;
    if (idx >= KB_ * KT_ * 128) return;
    int bt = idx >> 7, d = idx & 127;
    int t = bt & (KT_ - 1);
    float lr = -log1pf(expf(lam_re[d]));
    float li = lam_im[d];
    float dv = dt[t];
    float er = expf(dv * lr);
    float odr = er * cosf(dv * li), odi = er * sinf(dv * li);
    float nr = odr - 1.f, ni = odi;
    float den = lr * lr + li * li;
    float ofr = (nr * lr + ni * li) / den;
    float ofi = (ni * lr - nr * li) / den;
    odr_[idx] = odr; odi_[idx] = odi; ofr_[idx] = ofr; ofi_[idx] = ofi;
}

// h_t = opd_t * h_{t-1} + forcing_t * opf_t ; in-place (forc becomes uo)
__global__ __launch_bounds__(256) void k_tscan(float* __restrict__ forc,
                                               const float* __restrict__ odr,
                                               const float* __restrict__ odi,
                                               const float* __restrict__ ofr,
                                               const float* __restrict__ ofi) {
    int idx = blockIdx.x * blockDim.x + threadIdx.x;
    if (idx >= KB_ * 256 * 128) return;
    int d = idx & 127, sp = (idx >> 7) & 255, b = idx >> 15;
    float hr = 0.f, hi = 0.f;
    for (int t = 0; t < KT_; ++t) {
        int bt = b * KT_ + t;
        size_t m = (size_t)bt * 256 + sp;
        float fr = forc[m * 256 + d], fi = forc[m * 256 + 128 + d];
        int od = bt * 128 + d;
        float ur = fr * ofr[od] - fi * ofi[od];
        float ui = fr * ofi[od] + fi * ofr[od];
        float nhr = odr[od] * hr - odi[od] * hi + ur;
        float nhi = odr[od] * hi + odi[od] * hr + ui;
        hr = nhr; hi = nhi;
        forc[m * 256 + d] = hr;
        forc[m * 256 + 128 + d] = hi;
    }
}

__global__ __launch_bounds__(256) void k_router(const float* __restrict__ ft,
                                                const float* __restrict__ rW,
                                                float* __restrict__ gates) {
    int m = blockIdx.x * blockDim.x + threadIdx.x;
    if (m >= KM_) return;
    float a[KE_] = {0.f, 0.f, 0.f, 0.f};
    for (int k = 0; k < 256; ++k) {
        float v = ft[(size_t)m * 256 + k];
        a[0] += v * rW[k * 4 + 0]; a[1] += v * rW[k * 4 + 1];
        a[2] += v * rW[k * 4 + 2]; a[3] += v * rW[k * 4 + 3];
    }
    float mx = fmaxf(fmaxf(a[0], a[1]), fmaxf(a[2], a[3]));
    float s = 0.f;
    for (int e = 0; e < KE_; ++e) { a[e] = expf(a[e] - mx); s += a[e]; }
    float inv = 1.f / s;
    for (int e = 0; e < KE_; ++e) gates[(size_t)m * KE_ + e] = a[e] * inv;
}

__global__ __launch_bounds__(256) void k_zupdate(float* __restrict__ z,
                                                 const float* __restrict__ xon,
                                                 const float* __restrict__ dtok) {
    int i = blockIdx.x * blockDim.x + threadIdx.x;
    if (i < KM_ * 256) z[i] += xon[i] + dtok[i];
}

// ---------------------------------------------------------------- launcher
static inline size_t alup(size_t x) { return (x + 255) & ~(size_t)255; }

extern "C" void kernel_launch(void* const* d_in, const int* in_sizes, int n_in,
                              void* d_out, int out_size, void* d_ws, size_t ws_size,
                              hipStream_t stream) {
    (void)in_sizes; (void)n_in; (void)out_size;
    const float* x       = (const float*)d_in[0];
    const float* dt      = (const float*)d_in[1];
    const float* enc_Wre = (const float*)d_in[2];
    const float* enc_Wim = (const float*)d_in[3];
    const float* dec_W   = (const float*)d_in[4];
    const float* dec_b   = (const float*)d_in[5];
    const float* ln_sp_g = (const float*)d_in[6];
    const float* ln_sp_b = (const float*)d_in[7];
    const float* conv_W  = (const float*)d_in[8];
    const float* conv_b  = (const float*)d_in[9];
    const float* E_re    = (const float*)d_in[10];
    const float* E_im    = (const float*)d_in[11];
    const float* Dc_re   = (const float*)d_in[12];
    const float* Dc_im   = (const float*)d_in[13];
    const float* gate_W  = (const float*)d_in[14];
    const float* gate_b  = (const float*)d_in[15];
    const float* src_W   = (const float*)d_in[16];
    const float* src_b   = (const float*)d_in[17];
    const float* fdecay  = (const float*)d_in[18];
    const float* lam_re  = (const float*)d_in[19];
    const float* lam_im  = (const float*)d_in[20];
    const float* ln_t_g  = (const float*)d_in[21];
    const float* ln_t_b  = (const float*)d_in[22];
    const float* rtr_W   = (const float*)d_in[23];
    const float* W1      = (const float*)d_in[24];
    const float* b1      = (const float*)d_in[25];
    const float* W2      = (const float*)d_in[26];
    const float* b2      = (const float*)d_in[27];
    float* out = (float*)d_out;

    char* w = (char*)d_ws;
    size_t off = 0;
    auto take = [&](size_t bytes) { char* p = w + off; off = alup(off + bytes); return p; };
    _Float16* SA   = (_Float16*)take((size_t)KM_ * 256 * 2);
    _Float16* SB   = (_Float16*)take((size_t)KM_ * KHD * 2);   // fused MoE hidden (f16)
    float* Z    = (float*)take((size_t)KM_ * 256 * 4);
    float* XE   = (float*)take((size_t)KM_ * 256 * 4);
    float* FORC = (float*)take((size_t)KM_ * 256 * 4);
    float* XO   = (float*)take((size_t)KM_ * 256 * 4);
    float* XON  = (float*)take((size_t)KM_ * 256 * 4);
    float* DTOK = (float*)take((size_t)KM_ * 256 * 4);
    float* XM   = (float*)take(16 * 256 * 4);
    float* FSEQ = (float*)take(16 * 256 * 4);
    float* SSEQ = (float*)take(16 * 256 * 4);
    float* GB   = (float*)take(16 * 128 * 4);
    float* ODR  = (float*)take(16 * 128 * 4);
    float* ODI  = (float*)take(16 * 128 * 4);
    float* OFR  = (float*)take(16 * 128 * 4);
    float* OFI  = (float*)take(16 * 128 * 4);
    float* FLUX = (float*)take(2 * 256 * 4);
    float* GATES = (float*)take((size_t)KM_ * 4 * 4);
    _Float16* WENC  = (_Float16*)take(256 * 256 * 2);
    _Float16* WDEC  = (_Float16*)take(256 * 256 * 2);
    _Float16* WCONV = (_Float16*)take((size_t)9 * 256 * 256 * 2);
    _Float16* WE    = (_Float16*)take(256 * 256 * 2);
    _Float16* WDC   = (_Float16*)take(256 * 256 * 2);
    _Float16* W1T   = (_Float16*)take((size_t)4 * 256 * 256 * 2);
    _Float16* W2T   = (_Float16*)take((size_t)4 * 256 * 256 * 2);
    if (off > ws_size) return;  // workspace too small: do nothing (deterministic)

    const int EW = KM_ * 256;
    const dim3 blk(256);
    const dim3 gEW((EW + 255) / 256);
    // GEMM grids: waves = (M/32)*(N/64); 8 waves per block
    const dim3 gGEMM(((KM_ / 32) * (KF2 / 64)) / 8);        // 4096x256  -> 64 blocks
    const dim3 gGEMM_MOE1(((KM_ / 32) * (KHD / 64)) / 8);   // 4096x1024 -> 256 blocks
    const dim3 gCONV(((KM_ / 16) * (KF2 / 64)) / 8);        // 1024 waves -> 128 blocks
    const dim3 gW64(((256 * 256) + 255) / 256);
    const dim3 gConvW(((9 * 256 * 256) + 255) / 256);

    // shared weight prep + state init
    k_prep_cat2T<<<gW64, blk, 0, stream>>>(enc_Wre, enc_Wim, WENC, 256, 128, 128);
    k_prep_T<<<gW64, blk, 0, stream>>>(dec_W, WDEC, 256, 256);
    k_fill0<<<dim3(2), blk, 0, stream>>>(FLUX, 512);

    // encoder: patchify -> f16, then z = f @ [Wre|Wim]
    k_patchify<<<gEW, blk, 0, stream>>>(x, SA);
    k_gemm<<<gGEMM, blk, 0, stream>>>(SA, WENC, Z, (_Float16*)nullptr,
                                      (const float*)nullptr, (const float*)nullptr,
                                      0, KM_, KF2, KF2, KF2, EP_STORE);

    for (int i = 0; i < 2; ++i) {
        // per-depth weight prep
        k_prep_conv<<<gConvW, blk, 0, stream>>>(conv_W + (size_t)i * 256 * 256 * 9, WCONV);
        k_prep_complexT<<<gW64, blk, 0, stream>>>(E_re + (size_t)i * 128 * 128,
                                                  E_im + (size_t)i * 128 * 128, WE);
        k_prep_complexT<<<gW64, blk, 0, stream>>>(Dc_re + (size_t)i * 128 * 128,
                                                  Dc_im + (size_t)i * 128 * 128, WDC);
        for (int e = 0; e < 4; ++e) {
            k_prep_T<<<gW64, blk, 0, stream>>>(W1 + ((size_t)(i * 4 + e)) * 256 * 256,
                                               W1T + (size_t)e * 256 * 256, 256, 256);
            k_prep_T<<<gW64, blk, 0, stream>>>(W2 + ((size_t)(i * 4 + e)) * 256 * 256,
                                               W2T + (size_t)e * 256 * 256, 256, 256);
        }
        // spatial LN -> f16, conv accumulates into Z
        k_layernorm<<<dim3(KM_), blk, 0, stream>>>(Z, ln_sp_g + i * 256, ln_sp_b + i * 256,
                                                   SA, (float*)nullptr);
        k_conv<<<gCONV, blk, 0, stream>>>(SA, WCONV, Z, conv_b + i * 256);
        // eigen encode
        k_f32_to_f16<<<gEW, blk, 0, stream>>>(Z, SA, EW);
        k_gemm<<<gGEMM, blk, 0, stream>>>(SA, WE, XE, (_Float16*)nullptr,
                                          (const float*)nullptr, (const float*)nullptr,
                                          0, KM_, KF2, KF2, KF2, EP_STORE);
        // flux tracker (recurrent) + gate
        k_mean<<<dim3(16), blk, 0, stream>>>(XE, XM);
        k_fluxscan<<<dim3(1), blk, 0, stream>>>(XM, gate_W + (size_t)i * 256 * 128,
                                                gate_b + i * 128, src_W + (size_t)i * 256 * 256,
                                                src_b + i * 256, fdecay + i * 128,
                                                FLUX, FSEQ, SSEQ, GB);
        k_forcing<<<gEW, blk, 0, stream>>>(XE, GB, SSEQ, FORC);
        // ZOH transition + time scan (FORC becomes uo)
        k_ops<<<dim3(8), blk, 0, stream>>>(dt, lam_re + i * 128, lam_im + i * 128,
                                           ODR, ODI, OFR, OFI);
        k_tscan<<<dim3(256), blk, 0, stream>>>(FORC, ODR, ODI, OFR, OFI);
        // basis decode + temporal LN
        k_f32_to_f16<<<gEW, blk, 0, stream>>>(FORC, SA, EW);
        k_gemm<<<gGEMM, blk, 0, stream>>>(SA, WDC, XO, (_Float16*)nullptr,
                                          (const float*)nullptr, (const float*)nullptr,
                                          0, KM_, KF2, KF2, KF2, EP_STORE);
        k_layernorm<<<dim3(KM_), blk, 0, stream>>>(XO, ln_t_g + i * 256, ln_t_b + i * 256,
                                                   SA, XON);  // SA = ft (f16), XON = ft (f32)
        // MoE
        k_router<<<dim3(16), blk, 0, stream>>>(XON, rtr_W + (size_t)i * 256 * 4, GATES);
        // stage 1 fused over experts: hidden[M][1024] = gelu(ft @ [W1T_e0|..|e3] + b1)
        k_gemm<<<gGEMM_MOE1, blk, 0, stream>>>(SA, W1T, (float*)nullptr, SB,
                                               b1 + (size_t)i * KHD,
                                               (const float*)nullptr, 0,
                                               KM_, KHD, KF2, KF2, EP_GELU_F16);
        // stage 2 per expert (A = expert slice of hidden, lda = 1024), gate-scaled accumulate
        for (int e = 0; e < 4; ++e) {
            k_gemm<<<gGEMM, blk, 0, stream>>>(SB + (size_t)e * 256, W2T + (size_t)e * 256 * 256,
                                              DTOK, (_Float16*)nullptr,
                                              b2 + (size_t)(i * 4 + e) * 256,
                                              GATES, e, KM_, KF2, KF2, KHD,
                                              (e == 0) ? EP_SCALE_STORE : EP_SCALE_ADD);
        }
        // residual: z += xcc + delta
        k_zupdate<<<gEW, blk, 0, stream>>>(Z, XON, DTOK);
    }

    // decoder: fo = z @ dec_W + dec_b, then unpatchify
    k_f32_to_f16<<<gEW, blk, 0, stream>>>(Z, SA, EW);
    k_gemm<<<gGEMM, blk, 0, stream>>>(SA, WDEC, XE, (_Float16*)nullptr, dec_b,
                                      (const float*)nullptr, 0, KM_, KF2, KF2, KF2, EP_STORE);
    k_unpatchify<<<gEW, blk, 0, stream>>>(XE, out);
}
